// TransformerLayer_52140902974218
// MI455X (gfx1250) — matrix operand order repaired and checked by reading, not compiled, not run on hardware
//
#include <hip/hip_runtime.h>

// ---------------------------------------------------------------------------
// Transformer decoder layer for MI455X (gfx1250, wave32, WMMA).
// B=2, L=2048, D=1024, H=8, DH=128.
// All GEMM-like ops: v_wmma_f32_16x16x32_bf16 with fp32 accumulation.
// B tiles (32x128 bf16) staged to LDS via GLOBAL_LOAD_ASYNC_TO_LDS_B128,
// double-buffered, ordered with s_wait_asynccnt + workgroup barrier.
// Each wave computes a 16x128 output strip (8 accumulators, 8 WMMA/k-step).
// ---------------------------------------------------------------------------

#define B_  2
#define L_  2048
#define D_  1024
#define H_  8
#define DH_ 128
#define M_  (B_ * L_)        // 4096 rows

#define BPITCH 136           // padded LDS row pitch (elements), 272 B (16B-mult)

typedef __bf16 bf16;
typedef __bf16 v16bf __attribute__((ext_vector_type(16)));
typedef float  v8f   __attribute__((ext_vector_type(8)));
typedef int    v4i   __attribute__((ext_vector_type(4)));

typedef __attribute__((address_space(1))) v4i* g_v4i_ptr;
typedef __attribute__((address_space(3))) v4i* l_v4i_ptr;

// ---------------------------------------------------------------------------
// CDNA5 async copy: global -> LDS, 16 bytes per call, tracked by ASYNCcnt.
// ---------------------------------------------------------------------------
__device__ __forceinline__ void async_copy_b128(const void* g, void* l) {
  // Low 32 bits of a flat LDS address are the raw LDS byte offset
  // (ISA 10.2); AS(3) pointers are 32-bit.
  unsigned loff = (unsigned)(uintptr_t)l;
#if __has_builtin(__builtin_amdgcn_global_load_async_to_lds_b128)
  __builtin_amdgcn_global_load_async_to_lds_b128(
      (g_v4i_ptr)(uintptr_t)g, (l_v4i_ptr)loff, /*offset=*/0, /*cpol=*/0);
#else
  unsigned long long ga = (unsigned long long)(uintptr_t)g;
  asm volatile("global_load_async_to_lds_b128 %0, %1, off"
               :: "v"(loff), "v"(ga) : "memory");
#endif
}

__device__ __forceinline__ void wait_async0() {
#if __has_builtin(__builtin_amdgcn_s_wait_asynccnt)
  __builtin_amdgcn_s_wait_asynccnt(0);
#else
  asm volatile("s_wait_asynccnt 0x0" ::: "memory");
#endif
}

// Stage one 32(K) x 128(N) bf16 tile (8 KB) into LDS (padded pitch BPITCH).
// 128 threads, each issues 4 async b128 copies (64 B).
__device__ __forceinline__ void stage_b_tile_async(const bf16* __restrict__ src,
                                                   size_t ldb,
                                                   bf16 (*dst)[BPITCH],
                                                   int tid) {
  const int r = tid >> 2;             // 0..31 (K)
  const int c = (tid & 3) * 32;       // 0,32,64,96 (N)
  const bf16* g = src + (size_t)r * ldb + c;
  async_copy_b128(g,      &dst[r][c]);
  async_copy_b128(g + 8,  &dst[r][c + 8]);
  async_copy_b128(g + 16, &dst[r][c + 16]);
  async_copy_b128(g + 24, &dst[r][c + 24]);
}

// ---------------------------------------------------------------------------
// WMMA helpers
// ---------------------------------------------------------------------------
__device__ __forceinline__ v8f wmma_bf16(v16bf a, v16bf b, v8f c) {
  return __builtin_amdgcn_wmma_f32_16x16x32_bf16(
      false, a, false, b, (short)0, c, false, false);
}

// A-fragment (16x32 bf16) from row-major fp32. Lanes 0-15: K 0..7 / 16..23 of
// row M=lane; lanes 16-31: K 8..15 / 24..31 of row M=lane-16.
__device__ __forceinline__ v16bf a_frag_from_f32(const float* __restrict__ p0,
                                                 int lda, int lane) {
  const float* p = p0 + (size_t)(lane & 15) * lda + ((lane >> 4) & 1) * 8;
  v16bf a;
#pragma unroll
  for (int i = 0; i < 8; ++i) a[i] = (bf16)p[i];
#pragma unroll
  for (int i = 0; i < 8; ++i) a[i + 8] = (bf16)p[i + 16];
  return a;
}

__device__ __forceinline__ void zero_acc8(v8f* acc) {
  for (int j = 0; j < 8; ++j)
#pragma unroll
    for (int i = 0; i < 8; ++i) acc[j][i] = 0.0f;
}

// ---------------------------------------------------------------------------
// GEMM: C[M,N] (fp32) = A[M,K] (fp32->bf16) x Bw[K,N] (bf16) (+bias)(+relu)
// 128 threads / 4 waves; block tile 64(M) x 128(N); double-buffered async B.
// ---------------------------------------------------------------------------
__global__ __launch_bounds__(128) void gemm_bf16_wmma(
    const float* __restrict__ A, const bf16* __restrict__ Bw,
    const float* __restrict__ bias, float* __restrict__ C,
    int Ndim, int Kdim, int relu) {
  __shared__ bf16 Bl[2][32][BPITCH];

  const int tid  = threadIdx.x;
  const int lane = tid & 31;
  const int wave = tid >> 5;
  const int n0   = blockIdx.x * 128;
  const int m0   = blockIdx.y * 64;
  const int mw   = m0 + wave * 16;

  v8f acc[8];
  zero_acc8(acc);

  const int nt = Kdim / 32;
  stage_b_tile_async(Bw + n0, Ndim, Bl[0], tid);

  for (int t = 0; t < nt; ++t) {
    const int cur = t & 1;
    wait_async0();          // this wave's tile-t copies landed
    __syncthreads();        // all waves: tile t ready, buf cur^1 free
    if (t + 1 < nt)
      stage_b_tile_async(Bw + (size_t)(t + 1) * 32 * Ndim + n0, Ndim,
                         Bl[cur ^ 1], tid);

    v16bf a = a_frag_from_f32(A + (size_t)mw * Kdim + t * 32, Kdim, lane);
#pragma unroll
    for (int j = 0; j < 8; ++j) {
      v16bf b;
#pragma unroll
      for (int i = 0; i < 16; ++i) b[i] = Bl[cur][(i & 7) + ((lane >> 4) & 1) * 8 + (i >> 3) * 16][j * 16 + (lane & 15)];
      acc[j] = wmma_bf16(a, b, acc[j]);
    }
  }

  const int mlo = (lane & 16) ? 8 : 0;
  const int nc  = lane & 15;
#pragma unroll
  for (int j = 0; j < 8; ++j) {
    const int n  = n0 + j * 16 + nc;
    const float bv = bias ? bias[n] : 0.0f;
#pragma unroll
    for (int r = 0; r < 8; ++r) {
      const int m = mw + mlo + r;
      float val = acc[j][r] + bv;
      if (relu) val = fmaxf(val, 0.0f);
      C[(size_t)m * Ndim + n] = val;
    }
  }
}

// ---------------------------------------------------------------------------
// Scores: S[bh,q,k] = (Q Kt)/sqrt(DH), optional causal mask.
// Q: [B*L, D] fp32 ; Kt: [B*H, DH, L] bf16 ; S: [B*H, L, L] fp32.
// Block tile: 64(q) x 128(key); K-dim = DH = 128.
// ---------------------------------------------------------------------------
__global__ __launch_bounds__(128) void attn_scores_wmma(
    const float* __restrict__ Q, const bf16* __restrict__ Kt,
    float* __restrict__ S, int causal) {
  const int tid  = threadIdx.x;
  const int lane = tid & 31;
  const int wave = tid >> 5;
  const int n0 = blockIdx.x * 128;    // key tile
  const int q0 = blockIdx.y * 64;     // query tile
  const int bh = blockIdx.z;
  if (causal && n0 > q0 + 63) return; // fully masked: skip (uniform)

  const int b  = bh / H_;
  const int h  = bh % H_;
  const int qw = q0 + wave * 16;

  const float* Qb = Q + (size_t)(b * L_ + qw) * D_ + h * DH_;
  const bf16*  Kb = Kt + (size_t)bh * DH_ * L_ + n0;

  __shared__ bf16 Bl[2][32][BPITCH];
  v8f acc[8];
  zero_acc8(acc);

  const int nt = DH_ / 32;            // 4
  stage_b_tile_async(Kb, L_, Bl[0], tid);

#pragma unroll
  for (int t = 0; t < nt; ++t) {
    const int cur = t & 1;
    wait_async0();
    __syncthreads();
    if (t + 1 < nt)
      stage_b_tile_async(Kb + (size_t)(t + 1) * 32 * L_, L_, Bl[cur ^ 1], tid);

    v16bf a = a_frag_from_f32(Qb + t * 32, D_, lane);
#pragma unroll
    for (int j = 0; j < 8; ++j) {
      v16bf bfrag;
#pragma unroll
      for (int i = 0; i < 16; ++i) bfrag[i] = Bl[cur][(i & 7) + ((lane >> 4) & 1) * 8 + (i >> 3) * 16][j * 16 + (lane & 15)];
      acc[j] = wmma_bf16(a, bfrag, acc[j]);
    }
  }

  const float scale = 0.08838834764831845f;  // 1/sqrt(128)
  const int mlo = (lane & 16) ? 8 : 0;
  const int nc  = lane & 15;
  float* Sb = S + (size_t)bh * L_ * L_;
#pragma unroll
  for (int j = 0; j < 8; ++j) {
    const int n = n0 + j * 16 + nc;
#pragma unroll
    for (int r = 0; r < 8; ++r) {
      const int qq = qw + mlo + r;
      float val = acc[j][r] * scale;
      if (causal && n > qq) val = -3.0e38f;
      Sb[(size_t)qq * L_ + n] = val;
    }
  }
}

// ---------------------------------------------------------------------------
// Row softmax over L=2048 (in place); masked tail written as zeros.
// ---------------------------------------------------------------------------
__global__ __launch_bounds__(256) void softmax_rows(float* __restrict__ S,
                                                    int causal) {
  const int row  = blockIdx.x;             // bh*L + q
  const int qpos = row & (L_ - 1);
  const int valid = causal ? (qpos + 1) : L_;
  float* p = S + (size_t)row * L_;
  const int t = threadIdx.x;

  float vals[8];
  float mx = -3.0e38f;
#pragma unroll
  for (int i = 0; i < 8; ++i) {
    const int idx = t + i * 256;
    const float x = (idx < valid) ? p[idx] : -3.0e38f;
    vals[i] = x;
    mx = fmaxf(mx, x);
  }

  __shared__ float red[256];
  red[t] = mx;
  __syncthreads();
  for (int s = 128; s > 0; s >>= 1) {
    if (t < s) red[t] = fmaxf(red[t], red[t + s]);
    __syncthreads();
  }
  mx = red[0];
  __syncthreads();

  float sum = 0.0f;
#pragma unroll
  for (int i = 0; i < 8; ++i) {
    const int idx = t + i * 256;
    vals[i] = (idx < valid) ? __expf(vals[i] - mx) : 0.0f;
    sum += vals[i];
  }
  red[t] = sum;
  __syncthreads();
  for (int s = 128; s > 0; s >>= 1) {
    if (t < s) red[t] += red[t + s];
    __syncthreads();
  }
  const float inv = 1.0f / red[0];
#pragma unroll
  for (int i = 0; i < 8; ++i) {
    const int idx = t + i * 256;
    p[idx] = vals[i] * inv;
  }
}

// ---------------------------------------------------------------------------
// O[b,q, h*DH+d] = sum_k P[bh,q,k] * V[b,k, h*DH+d]
// P: [B*H,L,L] fp32 (A side, cvt) ; Vb: [B*L,D] bf16 ; O: [B*L,D] fp32.
// Block tile: 64(q) x 128(d = full head); K-dim = L = 2048.
// ---------------------------------------------------------------------------
__global__ __launch_bounds__(128) void attn_av_wmma(
    const float* __restrict__ P, const bf16* __restrict__ Vb,
    float* __restrict__ O) {
  const int tid  = threadIdx.x;
  const int lane = tid & 31;
  const int wave = tid >> 5;
  const int q0 = blockIdx.x * 64;
  const int bh = blockIdx.y;
  const int b  = bh / H_;
  const int h  = bh % H_;
  const int qw = q0 + wave * 16;

  const float* Pb = P + ((size_t)bh * L_ + qw) * L_;
  const bf16*  Vbase = Vb + (size_t)(b * L_) * D_ + h * DH_;

  __shared__ bf16 Bl[2][32][BPITCH];
  v8f acc[8];
  zero_acc8(acc);

  const int nt = L_ / 32;             // 64
  stage_b_tile_async(Vbase, D_, Bl[0], tid);

  for (int t = 0; t < nt; ++t) {
    const int cur = t & 1;
    wait_async0();
    __syncthreads();
    if (t + 1 < nt)
      stage_b_tile_async(Vbase + (size_t)(t + 1) * 32 * D_, D_, Bl[cur ^ 1], tid);

    v16bf a = a_frag_from_f32(Pb + t * 32, L_, lane);
#pragma unroll
    for (int j = 0; j < 8; ++j) {
      v16bf bfrag;
#pragma unroll
      for (int i = 0; i < 16; ++i) bfrag[i] = Bl[cur][(i & 7) + ((lane >> 4) & 1) * 8 + (i >> 3) * 16][j * 16 + (lane & 15)];
      acc[j] = wmma_bf16(a, bfrag, acc[j]);
    }
  }

  const int mlo = (lane & 16) ? 8 : 0;
  const int nc  = lane & 15;
#pragma unroll
  for (int j = 0; j < 8; ++j) {
#pragma unroll
    for (int r = 0; r < 8; ++r) {
      const int qq = qw + mlo + r;
      O[(size_t)(b * L_ + qq) * D_ + h * DH_ + j * 16 + nc] = acc[j][r];
    }
  }
}

// ---------------------------------------------------------------------------
// Out = LayerNorm(Ain + Res) * g + beta    (rows of D=1024)
// ---------------------------------------------------------------------------
__global__ __launch_bounds__(256) void add_layernorm(
    const float* __restrict__ Ain, const float* __restrict__ Res,
    const float* __restrict__ g, const float* __restrict__ beta,
    float* __restrict__ Out) {
  const int row = blockIdx.x;
  const int t = threadIdx.x;
  const float* a = Ain + (size_t)row * D_;
  const float* r = Res + (size_t)row * D_;

  float x[4];
  float s = 0.0f, s2 = 0.0f;
#pragma unroll
  for (int i = 0; i < 4; ++i) {
    const int idx = t + i * 256;
    const float v = a[idx] + r[idx];
    x[i] = v;
    s += v;
    s2 += v * v;
  }
  __shared__ float rs[256];
  __shared__ float rs2[256];
  rs[t] = s; rs2[t] = s2;
  __syncthreads();
  for (int w = 128; w > 0; w >>= 1) {
    if (t < w) { rs[t] += rs[t + w]; rs2[t] += rs2[t + w]; }
    __syncthreads();
  }
  const float mean = rs[0] * (1.0f / (float)D_);
  const float var  = rs2[0] * (1.0f / (float)D_) - mean * mean;
  const float inv  = rsqrtf(var + 1e-6f);
#pragma unroll
  for (int i = 0; i < 4; ++i) {
    const int idx = t + i * 256;
    Out[(size_t)row * D_ + idx] = (x[i] - mean) * inv * g[idx] + beta[idx];
  }
}

// ---------------------------------------------------------------------------
// Conversions / layout transforms
// ---------------------------------------------------------------------------
__global__ __launch_bounds__(256) void cvt_weight_t(const float* __restrict__ W,
                                                    bf16* __restrict__ Wt) {
  const int idx = blockIdx.x * 256 + threadIdx.x;  // over D*D, out index
  const int n = idx & (D_ - 1);
  const int k = idx >> 10;
  Wt[idx] = (bf16)W[(size_t)n * D_ + k];
}

__global__ __launch_bounds__(256) void head_transpose_cvt(
    const float* __restrict__ K, bf16* __restrict__ Kt) {
  const int idx = blockIdx.x * 256 + threadIdx.x;  // over B*H*DH*L
  const int l = idx & (L_ - 1);
  const int rest = idx >> 11;
  const int d  = rest & (DH_ - 1);
  const int bh = rest >> 7;
  const int b  = bh >> 3;
  const int h  = bh & (H_ - 1);
  Kt[idx] = (bf16)K[(size_t)(b * L_ + l) * D_ + h * DH_ + d];
}

__global__ __launch_bounds__(256) void cvt_f32_bf16(const float* __restrict__ X,
                                                    bf16* __restrict__ Y) {
  const int idx = blockIdx.x * 256 + threadIdx.x;
  Y[idx] = (bf16)X[idx];
}

// ---------------------------------------------------------------------------
// Host-side orchestration
// ---------------------------------------------------------------------------
extern "C" void kernel_launch(void* const* d_in, const int* in_sizes, int n_in,
                              void* d_out, int out_size, void* d_ws,
                              size_t ws_size, hipStream_t stream) {
  (void)in_sizes; (void)n_in; (void)out_size; (void)ws_size;

  const float* x   = (const float*)d_in[0];
  const float* ctx = (const float*)d_in[1];
  const float* Wq  = (const float*)d_in[2];
  const float* Wk  = (const float*)d_in[3];
  const float* Wv  = (const float*)d_in[4];
  const float* lag = (const float*)d_in[5];
  const float* lab = (const float*)d_in[6];
  const float* w1  = (const float*)d_in[7];
  const float* b1  = (const float*)d_in[8];
  const float* w2  = (const float*)d_in[9];
  const float* b2  = (const float*)d_in[10];
  const float* lfg = (const float*)d_in[11];
  const float* lfb = (const float*)d_in[12];

  float* out  = (float*)d_out;
  float* attn = out + (size_t)B_ * L_ * D_;   // [B,H,L,L]; self-attn scratch too

  char* ws = (char*)d_ws;
  size_t off = 0;
  auto alloc = [&](size_t bytes) -> char* {
    char* p = ws + off;
    off += (bytes + 255) & ~(size_t)255;
    return p;
  };

  const size_t MD = (size_t)M_ * D_;
  float* q  = (float*)alloc(MD * 4);
  float* kk = (float*)alloc(MD * 4);
  float* v  = (float*)alloc(MD * 4);
  float* t0 = (float*)alloc(MD * 4);
  float* t1 = (float*)alloc(MD * 4);
  float* y1 = (float*)alloc(MD * 4);
  float* y2 = (float*)alloc(MD * 4);
  bf16* kt  = (bf16*)alloc(MD * 2);        // [B,H,DH,L]
  bf16* vb  = (bf16*)alloc(MD * 2);        // [B*L, D]
  bf16* wqt = (bf16*)alloc((size_t)D_ * D_ * 2);
  bf16* wkt = (bf16*)alloc((size_t)D_ * D_ * 2);
  bf16* wvt = (bf16*)alloc((size_t)D_ * D_ * 2);
  bf16* w1t = (bf16*)alloc((size_t)D_ * D_ * 2);
  bf16* w2t = (bf16*)alloc((size_t)D_ * D_ * 2);

  const dim3 blk128(128), blk256(256);
  const dim3 ggemm(D_ / 128, M_ / 64);               // (8, 64)
  const dim3 gscore(L_ / 128, L_ / 64, B_ * H_);     // (16, 32, 16)
  const dim3 gav(L_ / 64, B_ * H_);                  // (32, 16)
  const int wblk = (D_ * D_) / 256;
  const int mblk = (int)(MD / 256);

  cvt_weight_t<<<wblk, blk256, 0, stream>>>(Wq, wqt);
  cvt_weight_t<<<wblk, blk256, 0, stream>>>(Wk, wkt);
  cvt_weight_t<<<wblk, blk256, 0, stream>>>(Wv, wvt);
  cvt_weight_t<<<wblk, blk256, 0, stream>>>(w1, w1t);
  cvt_weight_t<<<wblk, blk256, 0, stream>>>(w2, w2t);

  // ---- causal self-attention ----
  gemm_bf16_wmma<<<ggemm, blk128, 0, stream>>>(x, wqt, nullptr, q, D_, D_, 0);
  gemm_bf16_wmma<<<ggemm, blk128, 0, stream>>>(x, wkt, nullptr, kk, D_, D_, 0);
  gemm_bf16_wmma<<<ggemm, blk128, 0, stream>>>(x, wvt, nullptr, v, D_, D_, 0);
  head_transpose_cvt<<<mblk, blk256, 0, stream>>>(kk, kt);
  cvt_f32_bf16<<<mblk, blk256, 0, stream>>>(v, vb);
  attn_scores_wmma<<<gscore, blk128, 0, stream>>>(q, kt, attn, 1);
  softmax_rows<<<B_ * H_ * L_, blk256, 0, stream>>>(attn, 1);
  attn_av_wmma<<<gav, blk128, 0, stream>>>(attn, vb, t0);
  add_layernorm<<<M_, blk256, 0, stream>>>(t0, x, lag, lab, y1);

  // ---- FFN 1 ----
  gemm_bf16_wmma<<<ggemm, blk128, 0, stream>>>(y1, w1t, b1, t0, D_, D_, 1);
  gemm_bf16_wmma<<<ggemm, blk128, 0, stream>>>(t0, w2t, b2, t1, D_, D_, 0);
  add_layernorm<<<M_, blk256, 0, stream>>>(t1, y1, lfg, lfb, y2);

  // ---- cross-attention (probabilities are a real output) ----
  gemm_bf16_wmma<<<ggemm, blk128, 0, stream>>>(y2, wqt, nullptr, q, D_, D_, 0);
  gemm_bf16_wmma<<<ggemm, blk128, 0, stream>>>(ctx, wkt, nullptr, kk, D_, D_, 0);
  gemm_bf16_wmma<<<ggemm, blk128, 0, stream>>>(ctx, wvt, nullptr, v, D_, D_, 0);
  head_transpose_cvt<<<mblk, blk256, 0, stream>>>(kk, kt);
  cvt_f32_bf16<<<mblk, blk256, 0, stream>>>(v, vb);
  attn_scores_wmma<<<gscore, blk128, 0, stream>>>(q, kt, attn, 0);
  softmax_rows<<<B_ * H_ * L_, blk256, 0, stream>>>(attn, 0);
  attn_av_wmma<<<gav, blk128, 0, stream>>>(attn, vb, t0);
  add_layernorm<<<M_, blk256, 0, stream>>>(t0, y2, lag, lab, y1);

  // ---- FFN 2 -> final output ----
  gemm_bf16_wmma<<<ggemm, blk128, 0, stream>>>(y1, w1t, b1, t0, D_, D_, 1);
  gemm_bf16_wmma<<<ggemm, blk128, 0, stream>>>(t0, w2t, b2, t1, D_, D_, 0);
  add_layernorm<<<M_, blk256, 0, stream>>>(t1, y1, lfg, lfb, out);
}
